// EntityBase_2551210573968
// MI455X (gfx1250) — matrix operand order repaired and checked by reading, not compiled, not run on hardware
//
#include <hip/hip_runtime.h>
#include <hip/hip_bf16.h>

typedef __attribute__((ext_vector_type(16))) __bf16 v16bf;
typedef __attribute__((ext_vector_type(8)))  __bf16 v8bf;
typedef __attribute__((ext_vector_type(8)))  float  v8f;

constexpr int kNE = 64, kED = 128, kNA = 16, kE = 512, kHD = 64, kR = 512;
constexpr int kBT = 2048;
constexpr int kQKV = 3 * kE;

// LDS layout (bytes), 208KB total, regions reused across phases:
constexpr int OFF_KS  = 0;        // ks  bf16[64][512] (P2W,P3R); xs f32[64][128] (P0W,P1R) overlays 0..32K
constexpr int OFF_X1S = 65536;    // x1s bf16[64][512] (P1W,P2R) -> ws/att/a2 after P2
constexpr int OFF_VT  = 131072;   // vt  bf16[512][64] (P2W,P4R)
constexpr int OFF_QS  = 196608;   // qs  bf16[16][512] (P2W,P3R)
constexpr int OFF_WS  = 65536;    // ws  bf16[8][16][64] 16K (P3W,P4R)
constexpr int OFF_ATT = 81920;    // att bf16[16][512] 16K (P4W,P5R)
constexpr int OFF_A2  = 98304;    // a2  bf16[16][512] 16K (P5W,P6R)
constexpr int LDS_BYTES = 212992;

// ---- WMMA 16x16 tile GEMM, MT M-tiles share one B tile (B-register reuse) ----
// A: bf16 row-major (LDS), B: row-major weight W[n][k] (global/L2 or LDS)
template<int K, int MT>
__device__ __forceinline__ void wmma_mt_bf16(const __bf16* A, int lda,
                                             const __bf16* B, int ldb,
                                             int lane, v8f c[MT]) {
  const int m   = lane & 15;
  const int kbA = (lane >> 4) << 3;   // A: lanes 0-15: K 0..7/16..23; lanes 16-31: 8..15/24..31
  const int kbB = (lane >> 4) << 3;   // B: same K order as A (col n = row n of W)
  #pragma unroll
  for (int k0 = 0; k0 < K; k0 += 32) {
    const __bf16* pb = B + m * ldb + k0 + kbB;
    v8bf b0 = *(const v8bf*)pb;
    v8bf b1 = *(const v8bf*)(pb + 16);
    v16bf b;
    #pragma unroll
    for (int i = 0; i < 8; ++i) { b[i] = b0[i]; b[8 + i] = b1[i]; }
    #pragma unroll
    for (int j = 0; j < MT; ++j) {
      const __bf16* pa = A + (j * 16 + m) * lda + k0 + kbA;
      v8bf alo = *(const v8bf*)pa;
      v8bf ahi = *(const v8bf*)(pa + 16);
      v16bf a;
      #pragma unroll
      for (int i = 0; i < 8; ++i) { a[i] = alo[i]; a[8 + i] = ahi[i]; }
      c[j] = __builtin_amdgcn_wmma_f32_16x16x32_bf16(false, a, false, b, (short)0,
                                                     c[j], false, false);
    }
  }
}

// ---- same, A is f32 (LDS-staged entities), converted in-register ----
template<int K, int MT>
__device__ __forceinline__ void wmma_mt_f32A(const float* A, int lda,
                                             const __bf16* B, int ldb,
                                             int lane, v8f c[MT]) {
  const int m   = lane & 15;
  const int kbA = (lane >> 4) << 3;
  const int kbB = (lane >> 4) << 3;
  #pragma unroll
  for (int k0 = 0; k0 < K; k0 += 32) {
    const __bf16* pb = B + m * ldb + k0 + kbB;
    v8bf b0 = *(const v8bf*)pb;
    v8bf b1 = *(const v8bf*)(pb + 16);
    v16bf b;
    #pragma unroll
    for (int i = 0; i < 8; ++i) { b[i] = b0[i]; b[8 + i] = b1[i]; }
    #pragma unroll
    for (int j = 0; j < MT; ++j) {
      const float* pa = A + (j * 16 + m) * lda + k0 + kbA;
      v8f alo = *(const v8f*)pa;
      v8f ahi = *(const v8f*)(pa + 16);
      v16bf a;
      #pragma unroll
      for (int i = 0; i < 8; ++i) { a[i] = (__bf16)alo[i]; a[8 + i] = (__bf16)ahi[i]; }
      c[j] = __builtin_amdgcn_wmma_f32_16x16x32_bf16(false, a, false, b, (short)0,
                                                     c[j], false, false);
    }
  }
}

__global__ void cvt_f32_to_bf16(const float* __restrict__ s,
                                __bf16* __restrict__ d, int n) {
  int i = blockIdx.x * blockDim.x + threadIdx.x;
  if (i < n) d[i] = (__bf16)s[i];
}

__global__ __launch_bounds__(256, 1)
void fused_entity_kernel(const float* __restrict__ ent,
                         const int*   __restrict__ obsm,
                         const int*   __restrict__ entm,
                         const __bf16* __restrict__ W1b,  const float* __restrict__ b1,
                         const __bf16* __restrict__ Winb,
                         const __bf16* __restrict__ Woutb, const float* __restrict__ bout,
                         const __bf16* __restrict__ W2b,   const float* __restrict__ b2,
                         float* __restrict__ out) {
  __shared__ __align__(16) char smem[LDS_BYTES];
  float*  xs  = (float*) (smem + OFF_KS);    // overlays ks until P2
  __bf16* ks  = (__bf16*)(smem + OFF_KS);
  __bf16* x1s = (__bf16*)(smem + OFF_X1S);
  __bf16* vt  = (__bf16*)(smem + OFF_VT);
  __bf16* qs  = (__bf16*)(smem + OFF_QS);
  __bf16* wsm = (__bf16*)(smem + OFF_WS);
  __bf16* att = (__bf16*)(smem + OFF_ATT);
  __bf16* a2  = (__bf16*)(smem + OFF_A2);

  const int bt   = blockIdx.x;
  const int tid  = threadIdx.x;
  const int lane = tid & 31;
  const int wv   = tid >> 5;                 // 8 waves
  const int n    = lane & 15;                // C/D: lane holds column n
  const int mbse = (lane >> 4) << 3;         // C/D: rows mbse..mbse+7

  const float* xg = ent + (size_t)bt * kNE * kED;

  // ---- P0: async-DMA entities (64x128 f32 = 32KB) into LDS ----
  {
    const unsigned lds0 = (unsigned)(uintptr_t)(smem + OFF_KS);
    const unsigned long long g0 = (unsigned long long)(uintptr_t)xg;
    #pragma unroll
    for (int it = 0; it < 8; ++it) {
      unsigned           ldsa = lds0 + (unsigned)(it * 4096 + tid * 16);
      unsigned long long ga   = g0 + (unsigned long long)(it * 4096 + tid * 16);
      asm volatile("global_load_async_to_lds_b128 %0, %1, off"
                   :: "v"(ldsa), "v"(ga) : "memory");
    }
    asm volatile("s_wait_asynccnt 0" ::: "memory");
  }
  __syncthreads();

  // ---- P1: x1 = relu(x @ W1^T + b1), 4 M-tiles per column job (B reuse) ----
  for (int t = wv; t < 32; t += 8) {
    v8f c[4] = {};
    wmma_mt_f32A<kED, 4>(xs, kED, W1b + t * 16 * kED, kED, lane, c);
    const float bias = b1[t * 16 + n];
    #pragma unroll
    for (int j = 0; j < 4; ++j)
      #pragma unroll
      for (int i = 0; i < 8; ++i) {
        float v = c[j][i] + bias;
        x1s[(j * 16 + mbse + i) * kE + t * 16 + n] = (__bf16)(v > 0.f ? v : 0.f);
      }
  }
  __syncthreads();

  // ---- P2: q/k/v projections; k & v use 4 M-tile B reuse ----
  for (int t = wv; t < 96; t += 8) {
    const int which = t >> 5;                // 0=k 1=v 2=q
    const int nt = t & 31;
    if (which == 0) {
      v8f c[4] = {};
      wmma_mt_bf16<kE, 4>(x1s, kE, Winb + (kE + nt * 16) * kE, kE, lane, c);
      #pragma unroll
      for (int j = 0; j < 4; ++j)
        #pragma unroll
        for (int i = 0; i < 8; ++i)
          ks[(j * 16 + mbse + i) * kE + nt * 16 + n] = (__bf16)c[j][i];
    } else if (which == 1) {                 // v, stored transposed [feat][entity]
      v8f c[4] = {};
      wmma_mt_bf16<kE, 4>(x1s, kE, Winb + (2 * kE + nt * 16) * kE, kE, lane, c);
      #pragma unroll
      for (int j = 0; j < 4; ++j) {
        v8bf vv;
        #pragma unroll
        for (int i = 0; i < 8; ++i) vv[i] = (__bf16)c[j][i];
        *(v8bf*)(vt + (nt * 16 + n) * kNE + j * 16 + mbse) = vv;
      }
    } else {                                 // q: agents only (rows 0..15)
      v8f c[1] = {};
      wmma_mt_bf16<kE, 1>(x1s, kE, Winb + (nt * 16) * kE, kE, lane, c);
      #pragma unroll
      for (int i = 0; i < 8; ++i)
        qs[(mbse + i) * kE + nt * 16 + n] = (__bf16)c[0][i];
    }
  }
  __syncthreads();

  // ---- P3: logits + masked softmax, one head per wave, in-register ----
  {
    const int h = wv;
    v8f c[4];
    #pragma unroll
    for (int nt = 0; nt < 4; ++nt) {
      v8f z[1] = {};
      wmma_mt_bf16<kHD, 1>(qs + h * kHD, kE, ks + nt * 16 * kE + h * kHD, kE, lane, z);
      c[nt] = z[0];
    }
    float l[4][8];
    #pragma unroll
    for (int nt = 0; nt < 4; ++nt)
      #pragma unroll
      for (int i = 0; i < 8; ++i) {
        const int m = mbse + i, e = nt * 16 + n;
        const int mk = obsm[(size_t)bt * kNE * kNE + m * kNE + e];
        l[nt][i] = c[nt][i] * 0.125f + (mk ? -1e30f : 0.0f);
      }
    #pragma unroll
    for (int i = 0; i < 8; ++i) {
      float mx = l[0][i];
      #pragma unroll
      for (int nt = 1; nt < 4; ++nt) mx = fmaxf(mx, l[nt][i]);
      #pragma unroll
      for (int off = 1; off <= 8; off <<= 1)     // reduce within 16-lane halves
        mx = fmaxf(mx, __shfl_xor(mx, off, 32));
      float e4[4], s = 0.f;
      #pragma unroll
      for (int nt = 0; nt < 4; ++nt) { e4[nt] = __expf(l[nt][i] - mx); s += e4[nt]; }
      #pragma unroll
      for (int off = 1; off <= 8; off <<= 1)
        s += __shfl_xor(s, off, 32);
      const float inv = (mx < -1e29f) ? 0.f : (1.f / s);
      const int m = mbse + i;
      #pragma unroll
      for (int nt = 0; nt < 4; ++nt)
        wsm[(h * 16 + m) * kNE + nt * 16 + n] = (__bf16)(e4[nt] * inv);
    }
  }
  __syncthreads();

  // ---- P4: attn = w @ v (vt transposed => contiguous B) ----
  for (int t = wv; t < 32; t += 8) {
    const int h = t >> 2, dt = t & 3;
    v8f c[1] = {};
    wmma_mt_bf16<kNE, 1>(wsm + h * 16 * kNE, kNE,
                         vt + (h * kHD + dt * 16) * kNE, kNE, lane, c);
    #pragma unroll
    for (int i = 0; i < 8; ++i)
      att[(mbse + i) * kE + h * kHD + dt * 16 + n] = (__bf16)c[0][i];
  }
  __syncthreads();

  // ---- P5: Wout projection + bias, agent-mask, relu ----
  for (int t = wv; t < 32; t += 8) {
    v8f c[1] = {};
    wmma_mt_bf16<kE, 1>(att, kE, Woutb + t * 16 * kE, kE, lane, c);
    const float bias = bout[t * 16 + n];
    #pragma unroll
    for (int i = 0; i < 8; ++i) {
      const int m = mbse + i;
      const int am = entm[(size_t)bt * kNE + m];
      float v = c[0][i] + bias;
      v = v > 0.f ? v : 0.f;
      a2[m * kE + t * 16 + n] = (__bf16)(am ? 0.f : v);
    }
  }
  __syncthreads();

  // ---- P6: W2 projection + b2 + relu -> global out ----
  for (int t = wv; t < 32; t += 8) {
    v8f c[1] = {};
    wmma_mt_bf16<kE, 1>(a2, kE, W2b + t * 16 * kE, kE, lane, c);
    const float bias = b2[t * 16 + n];
    #pragma unroll
    for (int i = 0; i < 8; ++i) {
      float v = c[0][i] + bias;
      out[(size_t)bt * kNA * kR + (mbse + i) * kR + t * 16 + n] = v > 0.f ? v : 0.f;
    }
  }
}

extern "C" void kernel_launch(void* const* d_in, const int* in_sizes, int n_in,
                              void* d_out, int out_size, void* d_ws, size_t ws_size,
                              hipStream_t stream) {
  const float* entities = (const float*)d_in[0];
  const int*   obs      = (const int*)d_in[1];
  const int*   emask    = (const int*)d_in[2];
  const float* W1       = (const float*)d_in[3];
  const float* b1       = (const float*)d_in[4];
  const float* Win      = (const float*)d_in[5];
  const float* Wout     = (const float*)d_in[6];
  const float* bo       = (const float*)d_in[7];
  const float* W2       = (const float*)d_in[8];
  const float* b2       = (const float*)d_in[9];
  float* out = (float*)d_out;

  __bf16* W1b   = (__bf16*)d_ws;                 // 512*128
  __bf16* Winb  = W1b + kE * kED;                // 1536*512
  __bf16* Woutb = Winb + kQKV * kE;              // 512*512
  __bf16* W2b   = Woutb + kE * kE;               // 512*512

  cvt_f32_to_bf16<<<(kE * kED + 255) / 256, 256, 0, stream>>>(W1, W1b, kE * kED);
  cvt_f32_to_bf16<<<(kQKV * kE + 255) / 256, 256, 0, stream>>>(Win, Winb, kQKV * kE);
  cvt_f32_to_bf16<<<(kE * kE + 255) / 256, 256, 0, stream>>>(Wout, Woutb, kE * kE);
  cvt_f32_to_bf16<<<(kE * kE + 255) / 256, 256, 0, stream>>>(W2, W2b, kE * kE);

  fused_entity_kernel<<<kBT, 256, 0, stream>>>(entities, obs, emask,
                                               W1b, b1, Winb, Woutb, bo, W2b, b2, out);
}
